// GraphECL_66941360276200
// MI455X (gfx1250) — matrix operand order repaired and checked by reading, not compiled, not run on hardware
//
#include <hip/hip_runtime.h>
#include <hip/hip_bf16.h>
#include <math.h>

// ---------------------------------------------------------------------------
// GraphECL loss on MI455X (gfx1250, wave32, WMMA).
// Heavy math (all dense GEMMs + the two 10000x10000x256 Gram-matrix row sums,
// ~112 GFLOP total) runs on v_wmma_f32_16x16x32_bf16 with f32 accumulation.
// z/q are unit-norm so bf16 inputs keep logits accurate ahead of exp().
// Gram operands (10 MB bf16) stay resident in the 192 MB L2 -> compute bound;
// the hot loop is software-pipelined (double-buffered fragments + sched
// barriers so the compiler cannot re-serialize load->wmma).
// ---------------------------------------------------------------------------

typedef __attribute__((ext_vector_type(16))) __bf16 v16bf;
typedef __attribute__((ext_vector_type(8)))  float  v8f;
typedef __attribute__((ext_vector_type(8)))  unsigned short us8;

#if defined(__has_builtin)
#if __has_builtin(__builtin_amdgcn_sched_barrier)
#define SCHED_FENCE() __builtin_amdgcn_sched_barrier(0)
#endif
#endif
#ifndef SCHED_FENCE
#define SCHED_FENCE() asm volatile("" ::: "memory")
#endif

union Frag {
    v16bf v;
    unsigned short u[16];
    us8 h[2];
};

__device__ __forceinline__ unsigned short f2bf(float f) {
    unsigned u = __builtin_bit_cast(unsigned, f);
    u += 0x7FFFu + ((u >> 16) & 1u);          // round to nearest even
    return (unsigned short)(u >> 16);
}

__device__ __forceinline__ v8f wmma_bf16(const Frag& a, const Frag& b, v8f c) {
    // (neg_a, A, neg_b, B, c_mod, C, reuse_a, reuse_b)
    return __builtin_amdgcn_wmma_f32_16x16x32_bf16(
        false, a.v, false, b.v, (short)0, c, false, false);
}

#define DQ 256   // hidden / output width (all GEMM N = 256)
#define LP 40    // padded LDS row stride (ushorts): distinct bank groups per
                 // half-wave b128 read, preserves 16B alignment

// ---------------------------------------------------------------------------
// degrees: deg_out[src]++, deg_in[dst]++
// ---------------------------------------------------------------------------
__global__ void degrees_kernel(const int* __restrict__ src, const int* __restrict__ dst,
                               float* __restrict__ degO, float* __restrict__ degI, int E) {
    int e = blockIdx.x * blockDim.x + threadIdx.x;
    if (e < E) {
        atomicAdd(degO + src[e], 1.0f);
        atomicAdd(degI + dst[e], 1.0f);
    }
}

// ---------------------------------------------------------------------------
// Tiled WMMA-bf16 GEMM: C[M x 256] = A[M x K] @ W[K x 256] (+bias, compile-time)
// 128 threads = 4 waves; block tile = 16(M) x 64(N); K stepped by 32.
// Double-buffered LDS (global loads staged in regs, 1 barrier per chunk).
// ---------------------------------------------------------------------------
template <bool HAS_BIAS>
__global__ __launch_bounds__(128) void gemm_wmma_bf16(
        const float* __restrict__ A, const float* __restrict__ W,
        const float* __restrict__ bias, float* __restrict__ C, int M, int K) {
    __shared__ unsigned short aT[2][16][LP];   // [buf][m][k]
    __shared__ unsigned short bTt[2][64][LP];  // [buf][n][k]  (transposed)

    const int tid  = threadIdx.x;        // 0..127
    const int wave = tid >> 5;           // 0..3
    const int lane = tid & 31;
    const int half = lane >> 4;          // 0/1
    const int l16  = lane & 15;
    const int ibase = blockIdx.y * 16;         // M tile
    const int nblk  = blockIdx.x * 64;         // block N base
    const int nsub  = nblk + wave * 16;        // wave N base

    const int aRow = tid >> 3;           // 0..15
    const int aK4  = (tid & 7) * 4;      // 0,4,...,28
    const float* aPtr = A + (size_t)(ibase + aRow) * K + aK4;

    v8f acc = {};
    const int nc = K >> 5;               // # 32-wide K chunks

    // prologue: fetch chunk 0 through registers into LDS buffer 0
    float4 fa = *(const float4*)aPtr;
    float4 fb[4];
#pragma unroll
    for (int rep = 0; rep < 4; ++rep) {
        int flat = rep * 512 + tid * 4;
        int kk = flat >> 6, nn = flat & 63;
        fb[rep] = *(const float4*)(W + (size_t)kk * DQ + nblk + nn);
    }
    {
        aT[0][aRow][aK4 + 0] = f2bf(fa.x);
        aT[0][aRow][aK4 + 1] = f2bf(fa.y);
        aT[0][aRow][aK4 + 2] = f2bf(fa.z);
        aT[0][aRow][aK4 + 3] = f2bf(fa.w);
#pragma unroll
        for (int rep = 0; rep < 4; ++rep) {
            int flat = rep * 512 + tid * 4;
            int kk = flat >> 6, nn = flat & 63;
            bTt[0][nn + 0][kk] = f2bf(fb[rep].x);
            bTt[0][nn + 1][kk] = f2bf(fb[rep].y);
            bTt[0][nn + 2][kk] = f2bf(fb[rep].z);
            bTt[0][nn + 3][kk] = f2bf(fb[rep].w);
        }
    }

    for (int c = 0; c < nc; ++c) {
        __syncthreads();                 // buffer (c&1) ready
        const int cur = c & 1;
        const bool more = (c + 1) < nc;
        if (more) {                      // issue next chunk's global loads now
            int kb = (c + 1) << 5;
            fa = *(const float4*)(aPtr + kb);
#pragma unroll
            for (int rep = 0; rep < 4; ++rep) {
                int flat = rep * 512 + tid * 4;
                int kk = flat >> 6, nn = flat & 63;
                fb[rep] = *(const float4*)(W + (size_t)(kb + kk) * DQ + nblk + nn);
            }
        }
        SCHED_FENCE();                   // keep loads issued above the WMMA
        // fragments: contiguous b128 LDS loads per ISA 7.12.2 layout
        Frag a, b;
        a.h[0] = *(const us8*)&aT[cur][l16][half * 8];
        a.h[1] = *(const us8*)&aT[cur][l16][16 + half * 8];
        b.h[0] = *(const us8*)&bTt[cur][wave * 16 + l16][half * 8];
        b.h[1] = *(const us8*)&bTt[cur][wave * 16 + l16][16 + half * 8];
        acc = wmma_bf16(a, b, acc);
        SCHED_FENCE();
        if (more) {                      // drain staged regs into other buffer
            int nxt = cur ^ 1;
            aT[nxt][aRow][aK4 + 0] = f2bf(fa.x);
            aT[nxt][aRow][aK4 + 1] = f2bf(fa.y);
            aT[nxt][aRow][aK4 + 2] = f2bf(fa.z);
            aT[nxt][aRow][aK4 + 3] = f2bf(fa.w);
#pragma unroll
            for (int rep = 0; rep < 4; ++rep) {
                int flat = rep * 512 + tid * 4;
                int kk = flat >> 6, nn = flat & 63;
                bTt[nxt][nn + 0][kk] = f2bf(fb[rep].x);
                bTt[nxt][nn + 1][kk] = f2bf(fb[rep].y);
                bTt[nxt][nn + 2][kk] = f2bf(fb[rep].z);
                bTt[nxt][nn + 3][kk] = f2bf(fb[rep].w);
            }
        }
    }

    // --- C layout: lane holds rows m = half*8 + r, column n = l16 ---
    const int n = nsub + l16;
    const float bias_v = HAS_BIAS ? bias[n] : 0.0f;
#pragma unroll
    for (int r = 0; r < 8; ++r) {
        int m = ibase + half * 8 + r;
        C[(size_t)m * DQ + n] = acc[r] + bias_v;
    }
}

// ---------------------------------------------------------------------------
// GCN aggregation: out[dst] += xw[src] * rsqrt(degO[src]) * rsqrt(degI[dst])
// ---------------------------------------------------------------------------
__global__ void scatter_gcn(const float* __restrict__ xw, const int* __restrict__ src,
                            const int* __restrict__ dst, const float* __restrict__ degO,
                            const float* __restrict__ degI, float* __restrict__ out) {
    int e = blockIdx.x;
    int c = threadIdx.x;
    int s = src[e], d = dst[e];
    float ne = rsqrtf(degO[s]) * rsqrtf(degI[d]);
    atomicAdd(out + (size_t)d * DQ + c, xw[(size_t)s * DQ + c] * ne);
}

__global__ void bias_act(float* __restrict__ x, const float* __restrict__ b,
                         int total, int relu) {
    int i = blockIdx.x * blockDim.x + threadIdx.x;
    if (i < total) {
        float v = x[i] + b[i & (DQ - 1)];
        if (relu) v = fmaxf(v, 0.0f);
        x[i] = v;
    }
}

// ---------------------------------------------------------------------------
// BatchNorm (training stats, biased var), then apply + relu
// ---------------------------------------------------------------------------
__global__ void bn_stats(const float* __restrict__ t, float* __restrict__ mu,
                         float* __restrict__ var, int n) {
    __shared__ float s1[256], s2[256];
    int c = blockIdx.x, tid = threadIdx.x;
    float a = 0.0f, b = 0.0f;
    for (int r = tid; r < n; r += 256) {
        float v = t[(size_t)r * DQ + c];
        a += v; b += v * v;
    }
    s1[tid] = a; s2[tid] = b; __syncthreads();
    for (int s = 128; s > 0; s >>= 1) {
        if (tid < s) { s1[tid] += s1[tid + s]; s2[tid] += s2[tid + s]; }
        __syncthreads();
    }
    if (tid == 0) {
        float m = s1[0] / n;
        mu[c] = m;
        var[c] = s2[0] / n - m * m;
    }
}

__global__ void bn_apply_relu(float* __restrict__ t, const float* __restrict__ mu,
                              const float* __restrict__ var, const float* __restrict__ gamma,
                              const float* __restrict__ beta, int total) {
    int i = blockIdx.x * blockDim.x + threadIdx.x;
    if (i < total) {
        int c = i & (DQ - 1);
        float v = (t[i] - mu[c]) * rsqrtf(var[c] + 1e-5f) * gamma[c] + beta[c];
        t[i] = fmaxf(v, 0.0f);
    }
}

// ---------------------------------------------------------------------------
// Row L2-normalize (in place) + emit bf16 copy. One block (256 thr) per row.
// ---------------------------------------------------------------------------
__global__ void normalize_rows(float* __restrict__ x, unsigned short* __restrict__ xb) {
    __shared__ float red[8];
    int row = blockIdx.x, tid = threadIdx.x;
    float v = x[(size_t)row * DQ + tid];
    float ss = v * v;
#pragma unroll
    for (int o = 16; o > 0; o >>= 1) ss += __shfl_xor(ss, o, 32);
    if ((tid & 31) == 0) red[tid >> 5] = ss;
    __syncthreads();
    if (tid == 0) {
        float t = 0.0f;
#pragma unroll
        for (int i = 0; i < 8; ++i) t += red[i];
        red[0] = t;
    }
    __syncthreads();
    float inv = 1.0f / fmaxf(sqrtf(red[0]), 1e-12f);
    float z = v * inv;
    x[(size_t)row * DQ + tid] = z;
    xb[(size_t)row * DQ + tid] = f2bf(z);
}

// ---------------------------------------------------------------------------
// Gram-matrix row sums (dominant term):
//   negs[i]  = sum_j exp(z_i . z_j / T)
//   negs2[i] = sum_j exp(z_i . q_j / T)
// 1 block per 16-row i-tile; A fragments register-resident; 8 waves split the
// j-tiles. B fragments double-buffered and pipelined across chunks AND across
// j-tiles; sched barriers pin the issue order so waits become partial.
// ---------------------------------------------------------------------------
__global__ __launch_bounds__(256) void negsum_kernel(
        const unsigned short* __restrict__ zb, const unsigned short* __restrict__ qb,
        float* __restrict__ negs, float* __restrict__ negs2, int nTiles) {
    __shared__ float red1[16], red2[16];
    const int tid  = threadIdx.x;       // 0..255
    const int wave = tid >> 5;          // 0..7
    const int lane = tid & 31;
    const int half = lane >> 4;
    const int l16  = lane & 15;
    const int ibase = blockIdx.x * 16;

    // Preload this tile's A fragments for all 8 K-chunks (K = 256)
    Frag af[8];
#pragma unroll
    for (int kc = 0; kc < 8; ++kc) {
        const unsigned short* p = zb + (size_t)(ibase + l16) * DQ + kc * 32;
        af[kc].h[0] = *(const us8*)(p + half * 16);
        af[kc].h[1] = *(const us8*)(p + half * 16 + 8);
    }

    float acc1[8] = {0, 0, 0, 0, 0, 0, 0, 0};
    float acc2[8] = {0, 0, 0, 0, 0, 0, 0, 0};

    // Double-buffered B fragments, pipelined across chunks and j-tiles.
    Frag bz[2], bq[2];
    size_t rowB = (size_t)(wave * 16 + l16) * DQ;   // first tile: jt = wave
    {
        const unsigned short* pz = zb + rowB + half * 16;
        bz[0].h[0] = *(const us8*)pz;
        bz[0].h[1] = *(const us8*)(pz + 8);
        const unsigned short* pq = qb + rowB + half * 16;
        bq[0].h[0] = *(const us8*)pq;
        bq[0].h[1] = *(const us8*)(pq + 8);
    }

    for (int jt = wave; jt < nTiles; jt += 8) {
        // next tile's row base (self if none; loads are benign re-reads)
        const size_t rowN = (jt + 8 < nTiles)
                          ? (size_t)((jt + 8) * 16 + l16) * DQ : rowB;
        // deep prefetch (+16 tiles) - stays inside workspace, no guard branch
        const size_t rowP = (size_t)((jt + 16) * 16 + l16) * DQ;
        __builtin_prefetch(zb + rowP, 0, 3);
        __builtin_prefetch(qb + rowP, 0, 3);

        v8f c1 = {}, c2 = {};
#pragma unroll
        for (int kc = 0; kc < 8; ++kc) {
            const int cur = kc & 1, nxt = cur ^ 1;
            // issue loads for the next stage before consuming current stage
            const size_t ro  = (kc < 7) ? rowB : rowN;
            const int    off = (kc < 7) ? (kc + 1) * 32 : 0;
            const unsigned short* pz = zb + ro + off + half * 16;
            bz[nxt].h[0] = *(const us8*)pz;
            bz[nxt].h[1] = *(const us8*)(pz + 8);
            const unsigned short* pq = qb + ro + off + half * 16;
            bq[nxt].h[0] = *(const us8*)pq;
            bq[nxt].h[1] = *(const us8*)(pq + 8);
            SCHED_FENCE();               // loads may not sink below this point

            c1 = wmma_bf16(af[kc], bz[cur], c1);
            c2 = wmma_bf16(af[kc], bq[cur], c2);
            SCHED_FENCE();               // WMMAs may not hoist above the loads
        }
#pragma unroll
        for (int r = 0; r < 8; ++r) {
            acc1[r] += __expf(c1[r] * 2.0f);   // 1/TEMP = 2
            acc2[r] += __expf(c2[r] * 2.0f);
        }
        // stage parity: 8 chunks -> buffer 0 again holds next tile's chunk 0
        rowB = rowN;
    }

    if (tid < 16) { red1[tid] = 0.0f; red2[tid] = 0.0f; }
    __syncthreads();
#pragma unroll
    for (int r = 0; r < 8; ++r) {
        atomicAdd(&red1[half * 8 + r], acc1[r]);
        atomicAdd(&red2[half * 8 + r], acc2[r]);
    }
    __syncthreads();
    if (tid < 16) {
        negs[ibase + tid]  = red1[tid];
        negs2[ibase + tid] = red2[tid];
    }
}

// ---------------------------------------------------------------------------
// Per-edge: pos += dot(z[src], q[dst]) / T (into dst); neg += log(...) (into dst)
// one wave per edge
// ---------------------------------------------------------------------------
__global__ void edge_kernel(const float* __restrict__ z, const float* __restrict__ q,
                            const int* __restrict__ src, const int* __restrict__ dst,
                            const float* __restrict__ negs, const float* __restrict__ negs2,
                            float* __restrict__ posA, float* __restrict__ negA, int E) {
    int wave = threadIdx.x >> 5;
    int lane = threadIdx.x & 31;
    int e = blockIdx.x * 8 + wave;
    if (e >= E) return;
    int s = src[e], d = dst[e];
    float dot = 0.0f;
#pragma unroll
    for (int i = 0; i < 8; ++i) {
        int c = lane + i * 32;
        dot += z[(size_t)s * DQ + c] * q[(size_t)d * DQ + c];
    }
#pragma unroll
    for (int o = 16; o > 0; o >>= 1) dot += __shfl_xor(dot, o, 32);
    if (lane == 0) {
        atomicAdd(posA + d, dot * 2.0f);                        // /TEMP
        atomicAdd(negA + d, logf(negs[d] + 1e-3f * negs2[s]));  // LAM = 1e-3
    }
}

// ---------------------------------------------------------------------------
// loss = mean_v [ (-pos[v] + LAMBDA_LOSS*neg[v]) / deg_in[v] ]
// ---------------------------------------------------------------------------
__global__ void loss_kernel(const float* __restrict__ posA, const float* __restrict__ negA,
                            const float* __restrict__ degI, float* __restrict__ out, int n) {
    __shared__ float red[256];
    int tid = threadIdx.x;
    float a = 0.0f;
    for (int v = blockIdx.x * 256 + tid; v < n; v += gridDim.x * 256) {
        a += (-posA[v] + 1.0f * negA[v]) / degI[v];
    }
    red[tid] = a; __syncthreads();
    for (int s = 128; s > 0; s >>= 1) {
        if (tid < s) red[tid] += red[tid + s];
        __syncthreads();
    }
    if (tid == 0) atomicAdd(out, red[0] / (float)n);
}

// ---------------------------------------------------------------------------
extern "C" void kernel_launch(void* const* d_in, const int* in_sizes, int n_in,
                              void* d_out, int out_size, void* d_ws, size_t ws_size,
                              hipStream_t stream) {
    const float* feat  = (const float*)d_in[0];
    const float* W1    = (const float*)d_in[1];
    const float* b1    = (const float*)d_in[2];
    const float* W2    = (const float*)d_in[3];
    const float* b2    = (const float*)d_in[4];
    const float* Wt1   = (const float*)d_in[5];
    const float* bt1   = (const float*)d_in[6];
    const float* gamma = (const float*)d_in[7];
    const float* beta  = (const float*)d_in[8];
    const float* Wt2   = (const float*)d_in[9];
    const float* bt2   = (const float*)d_in[10];
    const float* Wp    = (const float*)d_in[11];
    const float* bp    = (const float*)d_in[12];
    const int*   src   = (const int*)d_in[13];
    const int*   dst   = (const int*)d_in[14];

    const int IN = 512;
    const int N  = in_sizes[0] / IN;     // 10000 (multiple of 16)
    const int E  = in_sizes[13];         // 330000

    char* ws = (char*)d_ws;
    size_t off = 0;
    auto alloc = [&](size_t bytes) -> void* {
        void* p = ws + off;
        off = (off + bytes + 255) & ~(size_t)255;
        return p;
    };
    float* bufA = (float*)alloc((size_t)N * DQ * 4);   // xw scratch / projector p -> q
    float* bufB = (float*)alloc((size_t)N * DQ * 4);   // GCN h -> z
    float* bufC = (float*)alloc((size_t)N * DQ * 4);   // t / bn(t); later aliased by zb/qb
    float* bufD = (float*)alloc((size_t)N * DQ * 4);   // trans
    float* degO  = (float*)alloc((size_t)N * 4);
    float* degI  = (float*)alloc((size_t)N * 4);
    float* negs  = (float*)alloc((size_t)N * 4);
    float* negs2 = (float*)alloc((size_t)N * 4);
    float* posA  = (float*)alloc((size_t)N * 4);
    float* negA  = (float*)alloc((size_t)N * 4);
    float* mu    = (float*)alloc(DQ * 4);
    float* var   = (float*)alloc(DQ * 4);

    const dim3 gemmGrid(DQ / 64, N / 16);   // (4, 625)
    const int  gemmBlk = 128;
    const int  total = N * DQ;

    // 1) degrees
    hipMemsetAsync(degO, 0, (size_t)N * 4, stream);
    hipMemsetAsync(degI, 0, (size_t)N * 4, stream);
    hipMemsetAsync(posA, 0, (size_t)N * 4, stream);
    hipMemsetAsync(negA, 0, (size_t)N * 4, stream);
    hipMemsetAsync(d_out, 0, 4, stream);
    degrees_kernel<<<(E + 255) / 256, 256, 0, stream>>>(src, dst, degO, degI, E);

    // 2) GCN layer 1: xw1 = feat @ W1 ; h1 = relu(agg + b1)
    gemm_wmma_bf16<false><<<gemmGrid, gemmBlk, 0, stream>>>(feat, W1, nullptr, bufA, N, IN);
    hipMemsetAsync(bufB, 0, (size_t)total * 4, stream);
    scatter_gcn<<<E, DQ, 0, stream>>>(bufA, src, dst, degO, degI, bufB);
    bias_act<<<(total + 255) / 256, 256, 0, stream>>>(bufB, b1, total, 1);

    // 3) GCN layer 2: xw2 = h1 @ W2 ; h = agg + b2
    gemm_wmma_bf16<false><<<gemmGrid, gemmBlk, 0, stream>>>(bufB, W2, nullptr, bufA, N, DQ);
    hipMemsetAsync(bufB, 0, (size_t)total * 4, stream);
    scatter_gcn<<<E, DQ, 0, stream>>>(bufA, src, dst, degO, degI, bufB);
    bias_act<<<(total + 255) / 256, 256, 0, stream>>>(bufB, b2, total, 0);

    // 4) target encoder: t = feat @ Wt1 + bt1 ; BN(train) ; relu
    gemm_wmma_bf16<true><<<gemmGrid, gemmBlk, 0, stream>>>(feat, Wt1, bt1, bufC, N, IN);
    bn_stats<<<DQ, 256, 0, stream>>>(bufC, mu, var, N);
    bn_apply_relu<<<(total + 255) / 256, 256, 0, stream>>>(bufC, mu, var, gamma, beta, total);

    // 5) trans = bn_relu @ Wt2 + bt2 ; p = trans @ Wp + bp
    gemm_wmma_bf16<true><<<gemmGrid, gemmBlk, 0, stream>>>(bufC, Wt2, bt2, bufD, N, DQ);
    gemm_wmma_bf16<true><<<gemmGrid, gemmBlk, 0, stream>>>(bufD, Wp, bp, bufA, N, DQ);

    // 6) z = normalize(h) (bufB), q = normalize(p) (bufA); bf16 copies alias bufC (free)
    unsigned short* zb = (unsigned short*)bufC;
    unsigned short* qb = zb + (size_t)N * DQ;
    normalize_rows<<<N, DQ, 0, stream>>>(bufB, zb);
    normalize_rows<<<N, DQ, 0, stream>>>(bufA, qb);

    // 7) dense NxN similarity row sums (WMMA-dominant)
    negsum_kernel<<<N / 16, 256, 0, stream>>>(zb, qb, negs, negs2, N / 16);

    // 8) per-edge pos / neg accumulation, then mean loss
    edge_kernel<<<(E + 7) / 8, 256, 0, stream>>>(bufB, bufA, src, dst, negs, negs2,
                                                 posA, negA, E);
    loss_kernel<<<64, 256, 0, stream>>>(posA, negA, degI, (float*)d_out, N);
}
